// MOTMPNet_68195490726274
// MI455X (gfx1250) — matrix operand order repaired and checked by reading, not compiled, not run on hardware
//
#include <hip/hip_runtime.h>
#include <hip/hip_bf16.h>

// ---------------------------------------------------------------------------
// MOT MPNet forward on gfx1250 (MI455X), fused-LDS version.
// All GEMMs run through v_wmma_f32_16x16x32_f16 (f16 in, f32 accum).
// Per-step edge/flow MLPs are fully fused (gather -> L1 -> LDS transpose ->
// L2 -> store/scatter) with weights resident in LDS; only enc_node (K=2048)
// and the update layer use the generic global-B WMMA GEMM.
// Input order: setup_inputs() insertion order, params flattened (W,b)/layer:
//  0 x_det[20000,2048] f32, 1 edge_attr[640000,6] f32, 2 edge_index[2,640000] i32
//  3..6 enc_node | 7..10 enc_edge | 11..14 edge_model | 15..18 flow_in
//  19..22 flow_out | 23..24 update | 25..28 cls_edge.  Output [3,640000,1] f32.
// ---------------------------------------------------------------------------

typedef _Float16 half_t;
typedef __attribute__((ext_vector_type(16))) _Float16 v16h;
typedef __attribute__((ext_vector_type(8)))  _Float16 v8h;
typedef __attribute__((ext_vector_type(8)))  float    v8f;

#define N_NODES 20000
#define N_EDGES 640000

__device__ inline v8h ldv8(const half_t* p) { return *(const v8h*)p; }
__device__ inline v16h mk_frag(v8h a0, v8h a1) {
  v16h a;
#pragma unroll
  for (int i = 0; i < 8; ++i) { a[i] = a0[i]; a[i + 8] = a1[i]; }
  return a;
}
__device__ inline v8f wmma32(v16h a, v16h b, v8f c) {
  return __builtin_amdgcn_wmma_f32_16x16x32_f16(false, a, false, b, (short)0, c,
                                                false, false);
}

// ---------------------------------------------------------------------------
// Generic WMMA GEMM (global B): C[M,Np](f16) = relu(A[M,Kp] @ W + b)
// Wt[n*Kp+k] = W[k][n], zero-padded. One wave per 16x16 tile.
// ---------------------------------------------------------------------------
__global__ __launch_bounds__(256) void wmma_gemm_kernel(
    const half_t* __restrict__ A, const half_t* __restrict__ Wt,
    const float* __restrict__ bias, half_t* __restrict__ C,
    half_t* __restrict__ C2, int M, int Kp, int Np)
{
  const int wave = threadIdx.x >> 5, lane = threadIdx.x & 31;
  const int tilesN = Np >> 4;
  const int tiles  = (M >> 4) * tilesN;
  const int tile   = blockIdx.x * 8 + wave;
  if (tile >= tiles) return;                 // wave-uniform
  const int tm = tile / tilesN, tn = tile % tilesN;
  const int r = lane & 15, hi = lane >> 4;
  const long arow = (long)(tm * 16 + r) * Kp;
  const long brow = (long)(tn * 16 + r) * Kp;
  const int kA = hi ? 8 : 0;   // A halfs 0-7 -> K kA.., halfs 8-15 -> K kA+16..
  const int kB = hi ? 8 : 0;   // B halfs 0-7 -> K kB.., halfs 8-15 -> K kB+16..

  v8f acc = {};
  for (int kc = 0; kc < Kp; kc += 32) {
    v16h a = mk_frag(ldv8(A + arow + kc + kA), ldv8(A + arow + kc + kA + 16));
    v16h b = mk_frag(ldv8(Wt + brow + kc + kB), ldv8(Wt + brow + kc + kB + 16));
    acc = wmma32(a, b, acc);
  }
  const int colg = tn * 16 + r;
  const float bc = bias[colg];
#pragma unroll
  for (int j = 0; j < 8; ++j) {
    const int rowg = tm * 16 + j + hi * 8;
    half_t h = (half_t)fmaxf(acc[j] + bc, 0.f);
    C[(long)rowg * Np + colg] = h;
    if (C2) C2[(long)rowg * Np + colg] = h;
  }
}

// ---------------------------------------------------------------------------
// Fused 2-layer edge MLP: KP1 -> 96(ReLU) -> 16(ReLU).
// GM==0: edge_model gather [init_n[row],h_n[row],init_n[col],h_n[col],init_e,h_e]
// GM==1: enc_edge, input = f32 edge_attr[M,6] zero-padded to K=32.
// Weights in LDS; hidden tile bounced through per-wave LDS staging.
// ---------------------------------------------------------------------------
template <int KP1, int GM>
__global__ __launch_bounds__(256) void fused_edge_kernel(
    const float* __restrict__ eattr,
    const half_t* __restrict__ init_n, const half_t* __restrict__ h_n,
    const half_t* __restrict__ init_e, const half_t* __restrict__ h_e_in,
    const int* __restrict__ row, const int* __restrict__ col,
    const half_t* __restrict__ gW1, const float* __restrict__ gB1,
    const half_t* __restrict__ gW2, const float* __restrict__ gB2,
    half_t* __restrict__ out, half_t* __restrict__ out2)
{
  constexpr int NP1 = 96, NP2 = 16, TPW = 4, NC1 = KP1 / 32;
  __shared__ alignas(16) half_t sW1[NP1 * KP1];
  __shared__ alignas(16) half_t sW2[NP2 * NP1];
  __shared__ float sB1[NP1];
  __shared__ float sB2[NP2];
  __shared__ alignas(16) half_t sStage[8][16 * NP1];

  for (int i = threadIdx.x; i < NP1 * KP1; i += 256) sW1[i] = gW1[i];
  for (int i = threadIdx.x; i < NP2 * NP1; i += 256) sW2[i] = gW2[i];
  if (threadIdx.x < NP1) sB1[threadIdx.x] = gB1[threadIdx.x];
  if (threadIdx.x < NP2) sB2[threadIdx.x] = gB2[threadIdx.x];
  __syncthreads();

  const int wave = threadIdx.x >> 5, lane = threadIdx.x & 31;
  const int r = lane & 15, hi = lane >> 4;
  const int kA = hi ? 8 : 0, kB = hi ? 8 : 0;
  half_t* stg = &sStage[wave][0];

  for (int t = 0; t < TPW; ++t) {
    const int tm = (blockIdx.x * 8 + wave) * TPW + t;
    const int ebase = tm * 16;
    const int e = ebase + r;

    int ridx = 0, cidx = 0;
    if constexpr (GM == 0) { ridx = row[e]; cidx = col[e]; }

    // ---- layer 1: gathered A fragments x LDS weights -> hidden [16x96] ----
    v8f acc[NP1 / 16];
#pragma unroll
    for (int i = 0; i < NP1 / 16; ++i) { v8f z = {}; acc[i] = z; }

#pragma unroll
    for (int kc = 0; kc < NC1; ++kc) {
      v16h a;
      if constexpr (GM == 0) {
        v8h a0, a1;
        if (kc == 0)      { a0 = ldv8(init_n + (long)ridx * 32 + kA); a1 = ldv8(init_n + (long)ridx * 32 + kA + 16); }
        else if (kc == 1) { a0 = ldv8(h_n    + (long)ridx * 32 + kA); a1 = ldv8(h_n    + (long)ridx * 32 + kA + 16); }
        else if (kc == 2) { a0 = ldv8(init_n + (long)cidx * 32 + kA); a1 = ldv8(init_n + (long)cidx * 32 + kA + 16); }
        else if (kc == 3) { a0 = ldv8(h_n    + (long)cidx * 32 + kA); a1 = ldv8(h_n    + (long)cidx * 32 + kA + 16); }
        else              { a0 = ldv8(init_e + (long)e * 16 + kA);    a1 = ldv8(h_e_in + (long)e * 16 + kA); }
        a = mk_frag(a0, a1);
      } else {
#pragma unroll
        for (int i = 0; i < 8; ++i) {
          int k0 = kA + i;
          a[i]     = (k0 < 6) ? (half_t)eattr[(long)e * 6 + k0] : (half_t)0.f;
          a[i + 8] = (half_t)0.f;   // K 16..31 padded
        }
      }
#pragma unroll
      for (int tn = 0; tn < NP1 / 16; ++tn) {
        v16h b = mk_frag(ldv8(&sW1[(tn * 16 + r) * KP1 + kc * 32 + kB]),
                         ldv8(&sW1[(tn * 16 + r) * KP1 + kc * 32 + kB + 16]));
        acc[tn] = wmma32(a, b, acc[tn]);
      }
    }

    // ---- epilogue 1: bias+ReLU -> per-wave LDS staging (row-major 16x96) ----
#pragma unroll
    for (int tn = 0; tn < NP1 / 16; ++tn) {
      float bb = sB1[tn * 16 + r];
#pragma unroll
      for (int j = 0; j < 8; ++j) {
        float v = fmaxf(acc[tn][j] + bb, 0.f);
        stg[(j + hi * 8) * NP1 + tn * 16 + r] = (half_t)v;
      }
    }

    // ---- layer 2: [16x96] @ [96x16] ----
    v8f acc2 = {};
#pragma unroll
    for (int kc = 0; kc < NP1 / 32; ++kc) {
      v16h a = mk_frag(ldv8(&stg[r * NP1 + kc * 32 + kA]),
                       ldv8(&stg[r * NP1 + kc * 32 + kA + 16]));
      v16h b = mk_frag(ldv8(&sW2[r * NP1 + kc * 32 + kB]),
                       ldv8(&sW2[r * NP1 + kc * 32 + kB + 16]));
      acc2 = wmma32(a, b, acc2);
    }
    const float b2 = sB2[r];
#pragma unroll
    for (int j = 0; j < 8; ++j) {
      const int e2 = ebase + j + hi * 8;
      half_t h = (half_t)fmaxf(acc2[j] + b2, 0.f);
      out[(long)e2 * 16 + r] = h;
      if (out2) out2[(long)e2 * 16 + r] = h;
    }
  }
}

// ---------------------------------------------------------------------------
// Fused flow kernel: msg gather (80, padded 96) -> 64(ReLU) -> 32(ReLU) for
// BOTH flow_out (m=0) and flow_in (m=1), then masked atomic scatter into
// A_upd[N,64] = [sum_in(0..31) | sum_out(32..63)].
// ---------------------------------------------------------------------------
__global__ __launch_bounds__(256) void fused_flow_kernel(
    const half_t* __restrict__ init_n, const half_t* __restrict__ h_n,
    const half_t* __restrict__ h_e,
    const int* __restrict__ row, const int* __restrict__ col,
    const half_t* __restrict__ gW1o, const float* __restrict__ gB1o,
    const half_t* __restrict__ gW2o, const float* __restrict__ gB2o,
    const half_t* __restrict__ gW1i, const float* __restrict__ gB1i,
    const half_t* __restrict__ gW2i, const float* __restrict__ gB2i,
    float* __restrict__ A_upd)
{
  constexpr int KP1 = 96, NP1 = 64, NP2 = 32, TPW = 4;
  __shared__ alignas(16) half_t sW1[2][NP1 * KP1];
  __shared__ alignas(16) half_t sW2[2][NP2 * NP1];
  __shared__ float sB1[2][NP1];
  __shared__ float sB2[2][NP2];
  __shared__ alignas(16) half_t sStage[8][16 * NP1];

  for (int i = threadIdx.x; i < NP1 * KP1; i += 256) { sW1[0][i] = gW1o[i]; sW1[1][i] = gW1i[i]; }
  for (int i = threadIdx.x; i < NP2 * NP1; i += 256) { sW2[0][i] = gW2o[i]; sW2[1][i] = gW2i[i]; }
  if (threadIdx.x < NP1) { sB1[0][threadIdx.x] = gB1o[threadIdx.x]; sB1[1][threadIdx.x] = gB1i[threadIdx.x]; }
  if (threadIdx.x < NP2) { sB2[0][threadIdx.x] = gB2o[threadIdx.x]; sB2[1][threadIdx.x] = gB2i[threadIdx.x]; }
  __syncthreads();

  const int wave = threadIdx.x >> 5, lane = threadIdx.x & 31;
  const int r = lane & 15, hi = lane >> 4;
  const int kA = hi ? 8 : 0, kB = hi ? 8 : 0;
  half_t* stg = &sStage[wave][0];

  for (int t = 0; t < TPW; ++t) {
    const int tm = (blockIdx.x * 8 + wave) * TPW + t;
    const int ebase = tm * 16;
    const int e = ebase + r;
    const int cidx = col[e];

    // gather msg fragments once: [init_n[col](32), h_n[col](32), h_e(16)+pad]
    v16h afr[3];
    afr[0] = mk_frag(ldv8(init_n + (long)cidx * 32 + kA),
                     ldv8(init_n + (long)cidx * 32 + kA + 16));
    afr[1] = mk_frag(ldv8(h_n + (long)cidx * 32 + kA),
                     ldv8(h_n + (long)cidx * 32 + kA + 16));
    { v8h z = {}; afr[2] = mk_frag(ldv8(h_e + (long)e * 16 + kA), z); }

#pragma unroll
    for (int m = 0; m < 2; ++m) {            // m=0: flow_out, m=1: flow_in
      v8f acc1[NP1 / 16];
#pragma unroll
      for (int i = 0; i < NP1 / 16; ++i) { v8f z = {}; acc1[i] = z; }
#pragma unroll
      for (int kc = 0; kc < KP1 / 32; ++kc)
#pragma unroll
        for (int tn = 0; tn < NP1 / 16; ++tn) {
          v16h b = mk_frag(ldv8(&sW1[m][(tn * 16 + r) * KP1 + kc * 32 + kB]),
                           ldv8(&sW1[m][(tn * 16 + r) * KP1 + kc * 32 + kB + 16]));
          acc1[tn] = wmma32(afr[kc], b, acc1[tn]);
        }

#pragma unroll
      for (int tn = 0; tn < NP1 / 16; ++tn) {
        float bb = sB1[m][tn * 16 + r];
#pragma unroll
        for (int j = 0; j < 8; ++j) {
          float v = fmaxf(acc1[tn][j] + bb, 0.f);
          stg[(j + hi * 8) * NP1 + tn * 16 + r] = (half_t)v;
        }
      }

      v8f acc2[NP2 / 16];
#pragma unroll
      for (int i = 0; i < NP2 / 16; ++i) { v8f z = {}; acc2[i] = z; }
#pragma unroll
      for (int kc = 0; kc < NP1 / 32; ++kc) {
        v16h a = mk_frag(ldv8(&stg[r * NP1 + kc * 32 + kA]),
                         ldv8(&stg[r * NP1 + kc * 32 + kA + 16]));
#pragma unroll
        for (int tn = 0; tn < NP2 / 16; ++tn) {
          v16h b = mk_frag(ldv8(&sW2[m][(tn * 16 + r) * NP1 + kc * 32 + kB]),
                           ldv8(&sW2[m][(tn * 16 + r) * NP1 + kc * 32 + kB + 16]));
          acc2[tn] = wmma32(a, b, acc2[tn]);
        }
      }

      // masked scatter (relu + bias applied here); out mask r<c, in mask r>c
      const int foff = (m == 0) ? 32 : 0;
#pragma unroll
      for (int j = 0; j < 8; ++j) {
        const int e2 = ebase + j + hi * 8;
        const int r2 = row[e2], c2 = col[e2];
        const bool take = (m == 0) ? (r2 < c2) : (r2 > c2);
        if (take) {
#pragma unroll
          for (int tn = 0; tn < NP2 / 16; ++tn) {
            float v = fmaxf(acc2[tn][j] + sB2[m][tn * 16 + r], 0.f);
            atomicAdd(A_upd + (long)r2 * 64 + foff + tn * 16 + r, v);
          }
        }
      }
    }
  }
}

// --------------------------- helpers / small kernels -----------------------
__global__ void cvt_f32_f16_kernel(const float* __restrict__ s,
                                   half_t* __restrict__ d, long n)
{
  long t = (long)blockIdx.x * blockDim.x + threadIdx.x;
  if (t < n) d[t] = (half_t)s[t];
}

__global__ void cvt_weight_kernel(const float* __restrict__ W,
                                  const float* __restrict__ b,
                                  half_t* __restrict__ Wt, float* __restrict__ bp,
                                  int K, int N, int Kp, int Np)
{
  long t = (long)blockIdx.x * blockDim.x + threadIdx.x;
  if (t >= (long)Np * Kp) return;
  int n = (int)(t / Kp), k = (int)(t % Kp);
  Wt[t] = (k < K && n < N) ? (half_t)W[(long)k * N + n] : (half_t)0.f;
  if (k == 0) bp[n] = (n < N) ? b[n] : 0.f;
}

__global__ void zero_f32_kernel(float* __restrict__ p, long n)
{
  long t = (long)blockIdx.x * blockDim.x + threadIdx.x;
  if (t < n) p[t] = 0.f;
}

__global__ void counts_kernel(const int* __restrict__ row, const int* __restrict__ col,
                              float* __restrict__ cin, float* __restrict__ cout, int M)
{
  int e = blockIdx.x * blockDim.x + threadIdx.x;
  if (e >= M) return;
  int r = row[e], c = col[e];
  if (r < c)      atomicAdd(cout + r, 1.f);
  else if (r > c) atomicAdd(cin + r, 1.f);
}

__global__ void divide_kernel(const float* __restrict__ A_upd,
                              const float* __restrict__ cin,
                              const float* __restrict__ cout,
                              half_t* __restrict__ A16, int Nn)
{
  long t = (long)blockIdx.x * blockDim.x + threadIdx.x;
  if (t >= (long)Nn * 64) return;
  int n = (int)(t >> 6), j = (int)(t & 63);
  float cnt = (j < 32) ? cin[n] : cout[n];
  A16[t] = (half_t)(A_upd[t] / fmaxf(cnt, 1.f));
}

__global__ void cls_kernel(const half_t* __restrict__ h_e,
                           const float* __restrict__ W0, const float* __restrict__ b0,
                           const float* __restrict__ W1, const float* __restrict__ b1,
                           float* __restrict__ out, int M)
{
  int e = blockIdx.x * blockDim.x + threadIdx.x;
  if (e >= M) return;
  float x[16];
#pragma unroll
  for (int k = 0; k < 16; ++k) x[k] = (float)h_e[(long)e * 16 + k];
  float h[8];
#pragma unroll
  for (int o = 0; o < 8; ++o) {
    float s = b0[o];
#pragma unroll
    for (int k = 0; k < 16; ++k) s += x[k] * W0[k * 8 + o];
    h[o] = fmaxf(s, 0.f);
  }
  float s = b1[0];
#pragma unroll
  for (int o = 0; o < 8; ++o) s += h[o] * W1[o];
  out[e] = s;
}

// ---------------------------------------------------------------------------
static inline char* wsalloc(void* ws, size_t& off, size_t bytes) {
  char* p = (char*)ws + off;
  off = (off + bytes + 255) & ~(size_t)255;
  return p;
}
static inline int nblk(long n, int bs) { return (int)((n + bs - 1) / bs); }

extern "C" void kernel_launch(void* const* d_in, const int* in_sizes, int n_in,
                              void* d_out, int out_size, void* d_ws, size_t ws_size,
                              hipStream_t stream) {
  (void)in_sizes; (void)n_in; (void)out_size; (void)ws_size;
  const int M = N_EDGES, Nn = N_NODES;

  const float* x_det = (const float*)d_in[0];
  const float* eattr = (const float*)d_in[1];
  const int*   row   = (const int*)d_in[2];
  const int*   col   = (const int*)d_in[2] + M;
  float* out = (float*)d_out;

  // ---- workspace ----
  size_t off = 0;
  half_t* xf16   = (half_t*)wsalloc(d_ws, off, (size_t)Nn * 2048 * 2);
  half_t* h1n    = (half_t*)wsalloc(d_ws, off, (size_t)Nn * 128 * 2);
  half_t* init_n = (half_t*)wsalloc(d_ws, off, (size_t)Nn * 32 * 2);
  half_t* h_n    = (half_t*)wsalloc(d_ws, off, (size_t)Nn * 32 * 2);
  half_t* init_e = (half_t*)wsalloc(d_ws, off, (size_t)M * 16 * 2);
  half_t* h_e    = (half_t*)wsalloc(d_ws, off, (size_t)M * 16 * 2);
  float*  A_upd  = (float*) wsalloc(d_ws, off, (size_t)Nn * 64 * 4);
  half_t* A16    = (half_t*)wsalloc(d_ws, off, (size_t)Nn * 64 * 2);
  float*  cnt    = (float*) wsalloc(d_ws, off, (size_t)Nn * 2 * 4);
  float*  cin = cnt, *cout_ = cnt + Nn;

  struct L { const float *W, *b; int K, N, Kp, Np; half_t* Wt; float* bp; };
  L en0 = {(const float*)d_in[3],  (const float*)d_in[4],  2048, 128, 2048, 128, 0, 0};
  L en1 = {(const float*)d_in[5],  (const float*)d_in[6],   128,  32,  128,  32, 0, 0};
  L ee0 = {(const float*)d_in[7],  (const float*)d_in[8],     6,  80,   32,  96, 0, 0};
  L ee1 = {(const float*)d_in[9],  (const float*)d_in[10],   80,  16,   96,  16, 0, 0};
  L em0 = {(const float*)d_in[11], (const float*)d_in[12],  160,  80,  160,  96, 0, 0};
  L em1 = {(const float*)d_in[13], (const float*)d_in[14],   80,  16,   96,  16, 0, 0};
  L fi0 = {(const float*)d_in[15], (const float*)d_in[16],   80,  56,   96,  64, 0, 0};
  L fi1 = {(const float*)d_in[17], (const float*)d_in[18],   56,  32,   64,  32, 0, 0};
  L fo0 = {(const float*)d_in[19], (const float*)d_in[20],   80,  56,   96,  64, 0, 0};
  L fo1 = {(const float*)d_in[21], (const float*)d_in[22],   56,  32,   64,  32, 0, 0};
  L up  = {(const float*)d_in[23], (const float*)d_in[24],   64,  32,   64,  32, 0, 0};
  L* Ls[11] = {&en0,&en1,&ee0,&ee1,&em0,&em1,&fi0,&fi1,&fo0,&fo1,&up};
  for (int i = 0; i < 11; ++i) {
    Ls[i]->Wt = (half_t*)wsalloc(d_ws, off, (size_t)Ls[i]->Np * Ls[i]->Kp * 2);
    Ls[i]->bp = (float*) wsalloc(d_ws, off, (size_t)Ls[i]->Np * 4);
  }
  const float* cW0 = (const float*)d_in[25];
  const float* cb0 = (const float*)d_in[26];
  const float* cW1 = (const float*)d_in[27];
  const float* cb1 = (const float*)d_in[28];

  // ---- weight conversion + counts ----
  for (int i = 0; i < 11; ++i) {
    long n = (long)Ls[i]->Np * Ls[i]->Kp;
    cvt_weight_kernel<<<nblk(n, 256), 256, 0, stream>>>(
        Ls[i]->W, Ls[i]->b, Ls[i]->Wt, Ls[i]->bp,
        Ls[i]->K, Ls[i]->N, Ls[i]->Kp, Ls[i]->Np);
  }
  zero_f32_kernel<<<nblk(2L * Nn, 256), 256, 0, stream>>>(cnt, 2L * Nn);
  counts_kernel<<<nblk(M, 256), 256, 0, stream>>>(row, col, cin, cout_, M);

  auto gemm = [&](const half_t* A, const L& l, half_t* C, half_t* C2, int rows) {
    int tiles = (rows >> 4) * (l.Np >> 4);
    wmma_gemm_kernel<<<nblk(tiles, 8), 256, 0, stream>>>(
        A, l.Wt, l.bp, C, C2, rows, l.Kp, l.Np);
  };

  // ---- encoders ----
  cvt_f32_f16_kernel<<<nblk((long)Nn * 2048, 256), 256, 0, stream>>>(
      x_det, xf16, (long)Nn * 2048);
  gemm(xf16, en0, h1n, nullptr, Nn);               // [Nn,128]
  gemm(h1n, en1, init_n, h_n, Nn);                 // [Nn,32] x2

  const int eblocks = (M / 16) / 32;               // 8 waves x 4 tiles per block
  fused_edge_kernel<32, 1><<<eblocks, 256, 0, stream>>>(
      eattr, nullptr, nullptr, nullptr, nullptr, nullptr, nullptr,
      ee0.Wt, ee0.bp, ee1.Wt, ee1.bp, init_e, h_e);

  // ---- message-passing steps ----
  for (int step = 1; step <= 4; ++step) {
    fused_edge_kernel<160, 0><<<eblocks, 256, 0, stream>>>(
        nullptr, init_n, h_n, init_e, h_e, row, col,
        em0.Wt, em0.bp, em1.Wt, em1.bp, h_e, nullptr);   // in-place h_e update

    zero_f32_kernel<<<nblk((long)Nn * 64, 256), 256, 0, stream>>>(A_upd, (long)Nn * 64);
    fused_flow_kernel<<<eblocks, 256, 0, stream>>>(
        init_n, h_n, h_e, row, col,
        fo0.Wt, fo0.bp, fo1.Wt, fo1.bp,
        fi0.Wt, fi0.bp, fi1.Wt, fi1.bp, A_upd);

    divide_kernel<<<nblk((long)Nn * 64, 256), 256, 0, stream>>>(
        A_upd, cin, cout_, A16, Nn);
    gemm(A16, up, h_n, nullptr, Nn);               // h_n = [Nn,32]

    if (step >= 2)
      cls_kernel<<<nblk(M, 256), 256, 0, stream>>>(
          h_e, cW0, cb0, cW1, cb1, out + (size_t)(step - 2) * M, M);
  }
}
